// MultiHeadAttention_43164421325525
// MI455X (gfx1250) — matrix operand order repaired and checked by reading, not compiled, not run on hardware
//
#include <hip/hip_runtime.h>
#include <hip/hip_bf16.h>

// ---------------------------------------------------------------------------
// MI455X (gfx1250) wave32 WMMA multi-head attention, bf16 compute / f32 accum
// Async-to-LDS staging of shared operands (ASYNCcnt path), double-buffered.
// Fragment loads batched ahead of WMMA quads to avoid per-WMMA dscnt stalls.
// ---------------------------------------------------------------------------

#define Bsz  2
#define Tseq 2048
#define Dm   1024
#define NH   16
#define HDm  64

typedef __attribute__((ext_vector_type(16))) __bf16 bf16x16;
typedef __attribute__((ext_vector_type(8)))  float  f32x8;

union FragBF { bf16x16 v; uint4 q[2]; };

__device__ __forceinline__ unsigned short f2bf(float f) {
    unsigned int u = __float_as_uint(f);
    u += 0x7FFFu + ((u >> 16) & 1u);       // round-to-nearest-even
    return (unsigned short)(u >> 16);
}

// ---- CDNA5 async global->LDS copy (16B per lane), tracked by ASYNCcnt ------
__device__ __forceinline__ void async_b128(unsigned short* lds_dst,
                                           const unsigned short* gsrc) {
    // generic shared pointer: low 32 bits == wave-relative LDS byte offset
    unsigned int lo = (unsigned int)(unsigned long long)lds_dst;
    asm volatile("global_load_async_to_lds_b128 %0, %1, off"
                 :: "v"(lo), "v"(gsrc) : "memory");
}
__device__ __forceinline__ void wait_async() {
    asm volatile("s_wait_asynccnt 0x0" ::: "memory");
}

// ---------------------------------------------------------------------------
// f32 -> bf16 elementwise conversion (vectorized 4x)
// ---------------------------------------------------------------------------
__global__ void __launch_bounds__(256)
cvt_bf16_kernel(const float4* __restrict__ x, ushort4* __restrict__ y, int n4) {
    int i = blockIdx.x * 256 + threadIdx.x;
    if (i < n4) {
        float4 f = x[i];
        ushort4 o;
        o.x = f2bf(f.x); o.y = f2bf(f.y); o.z = f2bf(f.z); o.w = f2bf(f.w);
        y[i] = o;
    }
}

// ---------------------------------------------------------------------------
// Block-cooperative GEMM: 8 waves share one 64-wide N slab of W (one head).
// Each wave computes a 16x64 tile of C = A(bf16) @ W(bf16)^T, f32 accumulate.
// W K-slabs (64 rows x 32 k) staged via async-to-LDS, double buffered.
// ---------------------------------------------------------------------------
__device__ __forceinline__ void gemm_block_16x64(const unsigned short* __restrict__ A, int lda, int m0,
                                                 const unsigned short* __restrict__ W, int ldw, int n0,
                                                 unsigned short Wlds[2][64 * 32],
                                                 f32x8 acc[4]) {
    const int tid  = threadIdx.x;
    const int lane = tid & 31;
    const int ln = lane & 15, hi = lane >> 4;
    // staging assignment: 64 rows x 32 ushort slab, 16B chunks
    const int wrow = tid >> 2, wch = (tid & 3) * 8;

    const unsigned short* ap = A + (size_t)(m0 + ln) * lda;

    // prime slab 0
    async_b128(&Wlds[0][wrow * 32 + wch], W + (size_t)(n0 + wrow) * ldw + wch);
    wait_async();
    __syncthreads();

    FragBF a;       // A-frag (16x32): e<8 -> K=hi*8+e ; e>=8 -> K=16+hi*8+(e-8)
    a.q[0] = *(const uint4*)(ap + hi * 8);
    a.q[1] = *(const uint4*)(ap + 16 + hi * 8);

    for (int k0 = 0; k0 < Dm; k0 += 32) {
        const int  cur  = (k0 >> 5) & 1;
        const bool more = (k0 + 32) < Dm;
        if (more) {
            async_b128(&Wlds[cur ^ 1][wrow * 32 + wch],
                       W + (size_t)(n0 + wrow) * ldw + k0 + 32 + wch);
        }
        FragBF an;
        if (more) {   // register double-buffer next A-frag
            an.q[0] = *(const uint4*)(ap + k0 + 32 + hi * 8);
            an.q[1] = *(const uint4*)(ap + k0 + 48 + hi * 8);
            __builtin_prefetch(ap + k0 + 128, 0, 1);
        }
        const unsigned short* ls = &Wlds[cur][0];
        FragBF bf[4];   // batch all B-frag loads, then WMMAs back-to-back
#pragma unroll
        for (int j = 0; j < 4; ++j) {
            const unsigned short* bp = ls + (j * 16 + ln) * 32 + hi * 8;
            bf[j].q[0] = *(const uint4*)bp;
            bf[j].q[1] = *(const uint4*)(bp + 16);
        }
#pragma unroll
        for (int j = 0; j < 4; ++j) {
            acc[j] = __builtin_amdgcn_wmma_f32_16x16x32_bf16(
                false, a.v, false, bf[j].v, (short)0, acc[j], false, false);
        }
        wait_async();
        __syncthreads();
        if (more) a = an;
    }
}

// ---------------------------------------------------------------------------
// Q/K/V projection. mode 0/1: RoPE -> outR as [B][H][T][64]
//                   mode 2:   V    -> outVT transposed [B][H][64][T]
// Block = 8 waves on 8 consecutive 16-row M tiles of the same head.
// ---------------------------------------------------------------------------
__global__ void __launch_bounds__(256)
proj_kernel(const unsigned short* __restrict__ X,
            const unsigned short* __restrict__ W,
            unsigned short* __restrict__ outR,
            unsigned short* __restrict__ outVT,
            int mode) {
    __shared__ unsigned short Wlds[2][64 * 32];
    const int wslot = threadIdx.x >> 5;
    const int lane  = threadIdx.x & 31;
    const int ln = lane & 15, hi = lane >> 4;
    const int h  = blockIdx.x & 15;                         // shared head slab
    const int m0 = ((blockIdx.x >> 4) * 8 + wslot) * 16;    // per-wave M tile
    const int n0 = h * 64;

    f32x8 acc[4] = {};
    gemm_block_16x64(X, Dm, m0, W, Dm, n0, Wlds, acc);

    const int b = m0 >> 11;             // T = 2048, tiles never cross batch
    if (mode < 2) {
        // RoPE: inv_freq[i] = 10000^(-i/32); pairs (hd, hd+32) = (acc0,acc2),(acc1,acc3)
        const float cexp = 0.28782313662425572f;   // ln(10000)/32
        const float if0 = __expf(-cexp * (float)ln);
        const float if1 = __expf(-cexp * (float)(ln + 16));
        unsigned short* obase = outR + (size_t)(b * NH + h) * Tseq * HDm;
#pragma unroll
        for (int r = 0; r < 8; ++r) {
            const int t = (m0 + r + 8 * hi) & (Tseq - 1);
            float sn0, cs0, sn1, cs1;
            sincosf((float)t * if0, &sn0, &cs0);
            sincosf((float)t * if1, &sn1, &cs1);
            const float x0 = acc[0][r], x1 = acc[1][r], x2 = acc[2][r], x3 = acc[3][r];
            unsigned short* op = obase + (size_t)t * HDm;
            op[ln]      = f2bf(x0 * cs0 - x2 * sn0);
            op[ln + 16] = f2bf(x1 * cs1 - x3 * sn1);
            op[ln + 32] = f2bf(x0 * sn0 + x2 * cs0);
            op[ln + 48] = f2bf(x1 * sn1 + x3 * cs1);
        }
    } else {
        // V stored transposed [B][H][64][T]: 8 consecutive t per lane -> b128 store
        const int t0 = m0 & (Tseq - 1);
#pragma unroll
        for (int j = 0; j < 4; ++j) {
            const int hd = j * 16 + ln;
            union { unsigned short s[8]; uint4 q; } pk;
#pragma unroll
            for (int r = 0; r < 8; ++r) pk.s[r] = f2bf(acc[j][r]);
            *(uint4*)(outVT + ((size_t)(b * NH + h) * HDm + hd) * Tseq + t0 + 8 * hi) = pk.q;
        }
    }
}

// ---------------------------------------------------------------------------
// Flash attention: block = 8 waves, all on the same (b,h); 8 q-tiles.
// K/V 32-key slabs async-staged to LDS, double buffered, shared by all waves.
// ---------------------------------------------------------------------------
__global__ void __launch_bounds__(256)
attn_kernel(const unsigned short* __restrict__ Q,
            const unsigned short* __restrict__ K,
            const unsigned short* __restrict__ VT,
            unsigned short* __restrict__ Aout) {
    __shared__ unsigned short Kl[2][32 * 64];   // [key][d]
    __shared__ unsigned short Vl[2][64 * 32];   // [d][t]   (V already transposed)
    __shared__ unsigned short Pl[8][16 * 32];   // per-wave P-tile reshape buffer
    const int tid   = threadIdx.x;
    const int wslot = tid >> 5;
    const int lane  = tid & 31;
    const int ln = lane & 15, hi = lane >> 4;
    const int qt = (blockIdx.x & 15) * 8 + wslot;
    const int h  = (blockIdx.x >> 4) & 15;
    const int b  = blockIdx.x >> 8;

    const unsigned short* qb = Q  + (size_t)(b * NH + h) * Tseq * HDm;
    const unsigned short* kb = K  + (size_t)(b * NH + h) * Tseq * HDm;
    const unsigned short* vb = VT + (size_t)(b * NH + h) * HDm * Tseq;
    unsigned short* pl = &Pl[wslot][0];

    // staging assignments (16B chunks)
    const int krow = tid >> 3, kch = (tid & 7) * 8;   // K slab: 32 rows x 64
    const int vrow = tid >> 2, vch = (tid & 3) * 8;   // V slab: 64 rows x 32

    // Q A-fragments for the two K=32 steps over HD=64
    FragBF qa[2];
    {
        const unsigned short* qp = qb + (size_t)(qt * 16 + ln) * HDm;
#pragma unroll
        for (int s = 0; s < 2; ++s) {
            qa[s].q[0] = *(const uint4*)(qp + s * 32 + hi * 8);
            qa[s].q[1] = *(const uint4*)(qp + s * 32 + 16 + hi * 8);
        }
    }

    // prime slab 0
    async_b128(&Kl[0][krow * 64 + kch], kb + (size_t)krow * HDm + kch);
    async_b128(&Vl[0][vrow * 32 + vch], vb + (size_t)vrow * Tseq + vch);
    wait_async();
    __syncthreads();

    float mrow[8], lrow[8];
#pragma unroll
    for (int r = 0; r < 8; ++r) { mrow[r] = -3.0e38f; lrow[r] = 0.0f; }
    f32x8 o[4] = {};

    for (int it = 0; it < Tseq / 32; ++it) {
        const int t0  = it * 32;
        const int cur = it & 1;
        if (t0 + 32 < Tseq) {   // issue next slabs; overlap with compute below
            async_b128(&Kl[cur ^ 1][krow * 64 + kch],
                       kb + (size_t)(t0 + 32 + krow) * HDm + kch);
            async_b128(&Vl[cur ^ 1][vrow * 32 + vch],
                       vb + (size_t)vrow * Tseq + t0 + 32 + vch);
        }
        // ---- scores S = (Q Kt) * 0.125 for keys [t0, t0+32) from LDS ----
        f32x8 s0 = {}, s1 = {};
        {
            const unsigned short* kl = &Kl[cur][0];
            FragBF kf[4];   // [tile(lo/hi keys)][k-step]
#pragma unroll
            for (int s = 0; s < 2; ++s) {
                const unsigned short* kp0 = kl + ln * 64 + s * 32 + hi * 8;
                kf[s].q[0] = *(const uint4*)kp0;
                kf[s].q[1] = *(const uint4*)(kp0 + 16);
                const unsigned short* kp1 = kl + (16 + ln) * 64 + s * 32 + hi * 8;
                kf[2 + s].q[0] = *(const uint4*)kp1;
                kf[2 + s].q[1] = *(const uint4*)(kp1 + 16);
            }
#pragma unroll
            for (int s = 0; s < 2; ++s) {
                s0 = __builtin_amdgcn_wmma_f32_16x16x32_bf16(
                    false, qa[s].v, false, kf[s].v, (short)0, s0, false, false);
                s1 = __builtin_amdgcn_wmma_f32_16x16x32_bf16(
                    false, qa[s].v, false, kf[2 + s].v, (short)0, s1, false, false);
            }
        }
        // ---- online softmax: rows live in 16-lane halves of the wave ----
        float alpha[8];
#pragma unroll
        for (int r = 0; r < 8; ++r) {
            float v0 = s0[r] * 0.125f;
            float v1 = s1[r] * 0.125f;
            float tm = fmaxf(v0, v1);
#pragma unroll
            for (int msk = 1; msk < 16; msk <<= 1) tm = fmaxf(tm, __shfl_xor(tm, msk, 32));
            float nm = fmaxf(mrow[r], tm);
            float a  = __expf(mrow[r] - nm);
            float p0 = __expf(v0 - nm);
            float p1 = __expf(v1 - nm);
            float rs = p0 + p1;
#pragma unroll
            for (int msk = 1; msk < 16; msk <<= 1) rs += __shfl_xor(rs, msk, 32);
            lrow[r] = lrow[r] * a + rs;
            mrow[r] = nm;
            alpha[r] = a;
            const int M = r + 8 * hi;
            pl[M * 32 + ln]      = f2bf(p0);
            pl[M * 32 + 16 + ln] = f2bf(p1);
        }
#pragma unroll
        for (int j = 0; j < 4; ++j)
#pragma unroll
            for (int r = 0; r < 8; ++r) o[j][r] *= alpha[r];

        // ---- reshape P through LDS (C-layout -> A-layout), then O += P @ V ----
        FragBF pa;
        {
            const unsigned short* pr = pl + ln * 32;
            pa.q[0] = *(const uint4*)(pr + hi * 8);
            pa.q[1] = *(const uint4*)(pr + 16 + hi * 8);
        }
        {
            const unsigned short* vl = &Vl[cur][0];
            FragBF vf[4];   // batch V-frag loads, then WMMAs back-to-back
#pragma unroll
            for (int j = 0; j < 4; ++j) {
                const unsigned short* vp = vl + (j * 16 + ln) * 32 + hi * 8;
                vf[j].q[0] = *(const uint4*)vp;
                vf[j].q[1] = *(const uint4*)(vp + 16);
            }
#pragma unroll
            for (int j = 0; j < 4; ++j) {
                o[j] = __builtin_amdgcn_wmma_f32_16x16x32_bf16(
                    false, pa.v, false, vf[j].v, (short)0, o[j], false, false);
            }
        }
        wait_async();
        __syncthreads();
    }

    // ---- normalize + merge heads into [B][T][D] bf16 ----
#pragma unroll
    for (int r = 0; r < 8; ++r) {
        const float inv = 1.0f / lrow[r];
        const int t = qt * 16 + r + 8 * hi;
        unsigned short* op = Aout + ((size_t)b * Tseq + t) * Dm + h * HDm;
#pragma unroll
        for (int j = 0; j < 4; ++j)
            op[j * 16 + ln] = f2bf(o[j][r] * inv);
    }
}

// ---------------------------------------------------------------------------
// Output projection: Y = A @ Wo^T + bo, f32 out
// ---------------------------------------------------------------------------
__global__ void __launch_bounds__(256)
outproj_kernel(const unsigned short* __restrict__ A,
               const unsigned short* __restrict__ W,
               const float* __restrict__ bo,
               float* __restrict__ out) {
    __shared__ unsigned short Wlds[2][64 * 32];
    const int wslot = threadIdx.x >> 5;
    const int lane  = threadIdx.x & 31;
    const int ln = lane & 15, hi = lane >> 4;
    const int n0 = (blockIdx.x & 15) * 64;
    const int m0 = ((blockIdx.x >> 4) * 8 + wslot) * 16;

    f32x8 acc[4] = {};
    gemm_block_16x64(A, Dm, m0, W, Dm, n0, Wlds, acc);

#pragma unroll
    for (int j = 0; j < 4; ++j) {
        const int col = n0 + j * 16 + ln;
        const float bias = bo[col];
#pragma unroll
        for (int r = 0; r < 8; ++r) {
            const int M = m0 + r + 8 * hi;
            out[(size_t)M * Dm + col] = acc[j][r] + bias;
        }
    }
}

// ---------------------------------------------------------------------------
extern "C" void kernel_launch(void* const* d_in, const int* in_sizes, int n_in,
                              void* d_out, int out_size, void* d_ws, size_t ws_size,
                              hipStream_t stream) {
    const float* q_in = (const float*)d_in[0];
    const float* k_in = (const float*)d_in[1];
    const float* v_in = (const float*)d_in[2];
    const float* Wq   = (const float*)d_in[3];
    const float* Wk   = (const float*)d_in[4];
    const float* Wv   = (const float*)d_in[5];
    const float* Wo   = (const float*)d_in[6];
    const float* bo   = (const float*)d_in[7];

    const size_t NX = (size_t)Bsz * Tseq * Dm;   // 4,194,304
    const size_t NW = (size_t)Dm * Dm;           // 1,048,576

    // workspace layout (bf16 as ushort): 7*NX + 4*NW elems = 64 MiB
    unsigned short* ws  = (unsigned short*)d_ws;
    unsigned short* xq  = ws;
    unsigned short* xk  = xq  + NX;
    unsigned short* xv  = xk  + NX;
    unsigned short* wqb = xv  + NX;
    unsigned short* wkb = wqb + NW;
    unsigned short* wvb = wkb + NW;
    unsigned short* wob = wvb + NW;
    unsigned short* qr  = wob + NW;   // RoPE'd Q  [B][H][T][64]
    unsigned short* kr  = qr  + NX;   // RoPE'd K  [B][H][T][64]
    unsigned short* vt  = kr  + NX;   // V^T       [B][H][64][T]
    unsigned short* ao  = vt  + NX;   // attention out [B][T][D]

    const int gx = (int)(NX / 4 / 256);   // 4096
    const int gw = (int)(NW / 4 / 256);   // 1024
    cvt_bf16_kernel<<<gx, 256, 0, stream>>>((const float4*)q_in, (ushort4*)xq, (int)(NX / 4));
    cvt_bf16_kernel<<<gx, 256, 0, stream>>>((const float4*)k_in, (ushort4*)xk, (int)(NX / 4));
    cvt_bf16_kernel<<<gx, 256, 0, stream>>>((const float4*)v_in, (ushort4*)xv, (int)(NX / 4));
    cvt_bf16_kernel<<<gw, 256, 0, stream>>>((const float4*)Wq, (ushort4*)wqb, (int)(NW / 4));
    cvt_bf16_kernel<<<gw, 256, 0, stream>>>((const float4*)Wk, (ushort4*)wkb, (int)(NW / 4));
    cvt_bf16_kernel<<<gw, 256, 0, stream>>>((const float4*)Wv, (ushort4*)wvb, (int)(NW / 4));
    cvt_bf16_kernel<<<gw, 256, 0, stream>>>((const float4*)Wo, (ushort4*)wob, (int)(NW / 4));

    // proj: 16 heads x 32 m-groups = 512 blocks, 8 waves each
    proj_kernel<<<512, 256, 0, stream>>>(xq, wqb, qr, nullptr, 0);
    proj_kernel<<<512, 256, 0, stream>>>(xk, wkb, kr, nullptr, 1);
    proj_kernel<<<512, 256, 0, stream>>>(xv, wvb, nullptr, vt, 2);

    // attn: (b=2) x (h=16) x (16 q-tile groups) = 512 blocks
    attn_kernel<<<512, 256, 0, stream>>>(qr, kr, vt, ao);

    outproj_kernel<<<512, 256, 0, stream>>>(ao, wob, bo, (float*)d_out);
}
